// SpecializedMultiHeadAttention_31275951850069
// MI455X (gfx1250) — compile-verified, matrix operand order checked by reading, not run on hardware
//
#include <hip/hip_runtime.h>
#include <hip/hip_bf16.h>

typedef _Float16 h16;
typedef __attribute__((ext_vector_type(16))) _Float16 v16h;
typedef __attribute__((ext_vector_type(8)))  float    v8f;

#define D_MODEL 1024
#define N_HEADS 16
#define N_SPECIAL 4
#define N_STD 12
#define D_K 64
#define BB 2
#define SS 2048
#define BS (BB*SS)          /* 4096 rows */
#define VOCAB 32000
#define NEGV (-1000000000.0f)

__device__ __forceinline__ v8f wmma_f16(v16h a, v16h b, v8f c) {
  return __builtin_amdgcn_wmma_f32_16x16x32_f16(false, a, false, b, (short)0, c, false, false);
}

// ---------------- conversion / prep kernels ----------------

__global__ void k_cvt_f32_to_f16(const float* __restrict__ in, h16* __restrict__ out, int n) {
  int i = blockIdx.x * blockDim.x + threadIdx.x;
  if (i < n) out[i] = (h16)in[i];
}

// wt[f][d] = w[d][f]  (so B-matrix columns become contiguous rows)
__global__ void k_cvt_w_transpose(const float* __restrict__ w, h16* __restrict__ wt) {
  int i = blockIdx.x * blockDim.x + threadIdx.x;           // i = d*1024 + f
  if (i < D_MODEL * D_MODEL) {
    int d = i >> 10, f = i & 1023;
    wt[f * D_MODEL + d] = (h16)w[i];
  }
}

// vt[((b*16+h)*64+dk)*2048 + s] = vh[(b*2048+s)*1024 + h*64+dk]
__global__ void k_build_vt(const h16* __restrict__ vh, h16* __restrict__ vt) {
  int i = blockIdx.x * blockDim.x + threadIdx.x;
  if (i < BB * N_HEADS * D_K * SS) {
    int s  = i & (SS - 1);
    int dk = (i >> 11) & 63;
    int h  = (i >> 17) & 15;
    int b  = i >> 21;
    vt[i] = vh[(size_t)(b * SS + s) * D_MODEL + h * D_K + dk];
  }
}

// maskb[(b*4+sp)*2048 + k] = vocab_member[sp][ids[b][k]] ? 0 : -1e9
__global__ void k_build_mask(const int* __restrict__ ids, const unsigned char* __restrict__ vm,
                             float* __restrict__ maskb) {
  int i = blockIdx.x * blockDim.x + threadIdx.x;
  if (i < BB * N_SPECIAL * SS) {
    int k  = i & (SS - 1);
    int sp = (i >> 11) & 3;
    int b  = i >> 13;
    int tok = ids[b * SS + k];
    maskb[i] = vm[sp * VOCAB + tok] ? 0.0f : NEGV;
  }
}

// ---------------- generic WMMA GEMM: Y[M x 1024] = A[M x 1024] * Bt^T + bias ----
// Each wave computes a 16x64 block (A-frag reused across 4 B-frags).
__global__ __launch_bounds__(128) void k_gemm_wmma(
    const h16* __restrict__ A, const h16* __restrict__ Bt,
    const float* __restrict__ bias,
    h16* __restrict__ out_h, float* __restrict__ out_f, int M)
{
  int wid  = (blockIdx.x * blockDim.x + threadIdx.x) >> 5;
  int lane = threadIdx.x & 31;
  int mtiles = M >> 4;
  int m0 = (wid % mtiles) * 16;
  int n0 = (wid / mtiles) * 64;
  int lm = lane & 15, lh = lane >> 4;

  v8f c0 = {}, c1 = {}, c2 = {}, c3 = {};
  const h16* ap  = A  + (size_t)(m0 + lm) * D_MODEL + lh * 16;
  const h16* b0p = Bt + (size_t)(n0 +  0 + lm) * D_MODEL + lh * 16;
  const h16* b1p = Bt + (size_t)(n0 + 16 + lm) * D_MODEL + lh * 16;
  const h16* b2p = Bt + (size_t)(n0 + 32 + lm) * D_MODEL + lh * 16;
  const h16* b3p = Bt + (size_t)(n0 + 48 + lm) * D_MODEL + lh * 16;

  for (int k = 0; k < D_MODEL; k += 32) {
    v16h a  = *(const v16h*)(ap  + k);
    v16h b0 = *(const v16h*)(b0p + k);
    v16h b1 = *(const v16h*)(b1p + k);
    v16h b2 = *(const v16h*)(b2p + k);
    v16h b3 = *(const v16h*)(b3p + k);
    c0 = wmma_f16(a, b0, c0);
    c1 = wmma_f16(a, b1, c1);
    c2 = wmma_f16(a, b2, c2);
    c3 = wmma_f16(a, b3, c3);
  }

  int n = lm;
  float bv0 = bias[n0 +  0 + n];
  float bv1 = bias[n0 + 16 + n];
  float bv2 = bias[n0 + 32 + n];
  float bv3 = bias[n0 + 48 + n];
  for (int r = 0; r < 8; r++) {
    int row = m0 + r + 8 * lh;
    size_t base = (size_t)row * D_MODEL + n0 + n;
    if (out_f) {
      out_f[base +  0] = c0[r] + bv0;
      out_f[base + 16] = c1[r] + bv1;
      out_f[base + 32] = c2[r] + bv2;
      out_f[base + 48] = c3[r] + bv3;
    } else {
      out_h[base +  0] = (h16)(c0[r] + bv0);
      out_h[base + 16] = (h16)(c1[r] + bv1);
      out_h[base + 32] = (h16)(c2[r] + bv2);
      out_h[base + 48] = (h16)(c3[r] + bv3);
    }
  }
}

// ---------------- fused attention: one block = (b, h, 16 query rows) ----------
// Phase 1: scores (16 x 2048) via WMMA into LDS (f32, scaled + masked)
// Phase 2: softmax rows in LDS; spec heads also write attn_spec to d_out
// Phase 3: O = P x V via WMMA (A-frags converted from LDS f32)
__global__ __launch_bounds__(128) void k_attention(
    const h16* __restrict__ qh, const h16* __restrict__ kh,
    const h16* __restrict__ vt, const float* __restrict__ maskb,
    float* __restrict__ attn_spec, h16* __restrict__ oh)
{
  __shared__ float sc[16 * SS];   // 128 KB (<= 320 KB/WGP on CDNA5)

  int bid  = blockIdx.x;
  int qt   = bid & 127;
  int h    = (bid >> 7) & 15;
  int b    = bid >> 11;
  int q0   = qt * 16;
  int tid  = threadIdx.x;
  int wave = tid >> 5;
  int lane = tid & 31;
  int lm = lane & 15, lh = lane >> 4;
  const float inv = 0.125f;              // 1/sqrt(64)
  bool spec = (h >= N_STD);
  int  hs   = h - N_STD;

  // Q fragments for this 16-row tile (A: 16x32, two K-halves of d_k=64)
  const h16* qp = qh + (size_t)(b * SS + q0 + lm) * D_MODEL + h * D_K + lh * 16;
  v16h aq0 = *(const v16h*)(qp);
  v16h aq1 = *(const v16h*)(qp + 32);

  const float* mrow = maskb + (size_t)(b * N_SPECIAL + (spec ? hs : 0)) * SS;

  // ---- Phase 1: scores ----
  for (int ct = wave; ct < 128; ct += 4) {
    int k0 = ct * 16;
    const h16* kp = kh + (size_t)(b * SS + k0 + lm) * D_MODEL + h * D_K + lh * 16;
    v16h bk0 = *(const v16h*)(kp);
    v16h bk1 = *(const v16h*)(kp + 32);
    v8f c = {};
    c = wmma_f16(aq0, bk0, c);
    c = wmma_f16(aq1, bk1, c);
    float mterm = spec ? mrow[k0 + lm] : 0.0f;
    for (int r = 0; r < 8; r++) {
      int m = r + 8 * lh;
      sc[m * SS + k0 + lm] = c[r] * inv + mterm;
    }
  }
  __syncthreads();

  // ---- Phase 2: softmax (4 rows per wave) ----
  for (int rr = 0; rr < 4; rr++) {
    int row = wave * 4 + rr;
    float* srow = sc + row * SS;
    float mx = -3.4e38f;
    for (int c = lane; c < SS; c += 32) mx = fmaxf(mx, srow[c]);
    for (int o = 16; o > 0; o >>= 1) mx = fmaxf(mx, __shfl_xor(mx, o, 32));
    float sum = 0.0f;
    for (int c = lane; c < SS; c += 32) {
      float e = __expf(srow[c] - mx);
      srow[c] = e;
      sum += e;
    }
    for (int o = 16; o > 0; o >>= 1) sum += __shfl_xor(sum, o, 32);
    float rs = 1.0f / sum;
    if (spec) {
      float* orow = attn_spec + ((size_t)(b * N_SPECIAL + hs) * SS + q0 + row) * SS;
      for (int c = lane; c < SS; c += 32) { float p = srow[c] * rs; srow[c] = p; orow[c] = p; }
    } else {
      for (int c = lane; c < SS; c += 32) srow[c] *= rs;
    }
  }
  __syncthreads();

  // ---- Phase 3: O = P x V (wave owns 16 of the 64 d_k columns) ----
  int dn = wave * 16;
  v8f c = {};
  const h16* vrow = vt + ((size_t)(b * N_HEADS + h) * D_K + dn + lm) * SS + lh * 16;
  for (int k = 0; k < SS; k += 32) {
    const float* pr = sc + lm * SS + k + lh * 16;
    v16h a;
    for (int i = 0; i < 16; i++) a[i] = (h16)pr[i];
    v16h bf = *(const v16h*)(vrow + k);
    c = wmma_f16(a, bf, c);
  }
  for (int r = 0; r < 8; r++) {
    int m = r + 8 * lh;
    oh[(size_t)(b * SS + q0 + m) * D_MODEL + h * D_K + dn + lm] = (h16)c[r];
  }
}

// ---------------- host launch ----------------

extern "C" void kernel_launch(void* const* d_in, const int* in_sizes, int n_in,
                              void* d_out, int out_size, void* d_ws, size_t ws_size,
                              hipStream_t stream) {
  const float* x   = (const float*)d_in[0];
  const float* wq  = (const float*)d_in[1];
  const float* bq  = (const float*)d_in[2];
  const float* wk  = (const float*)d_in[3];
  const float* bk  = (const float*)d_in[4];
  const float* wv  = (const float*)d_in[5];
  const float* bv  = (const float*)d_in[6];
  const float* wo  = (const float*)d_in[7];
  const float* bo  = (const float*)d_in[8];
  const int*   ids = (const int*)d_in[9];
  const unsigned char* vm = (const unsigned char*)d_in[10];

  float* final_out     = (float*)d_out;
  float* attn_spec_out = final_out + (size_t)BS * D_MODEL;

  char* ws = (char*)d_ws;
  size_t off = 0;
  auto alloc = [&](size_t bytes) -> char* {
    char* p = ws + off;
    off = (off + bytes + 255) & ~(size_t)255;
    return p;
  };
  h16* xh   = (h16*)alloc((size_t)BS * D_MODEL * 2);
  h16* wqt  = (h16*)alloc((size_t)D_MODEL * D_MODEL * 2);
  h16* wkt  = (h16*)alloc((size_t)D_MODEL * D_MODEL * 2);
  h16* wvt  = (h16*)alloc((size_t)D_MODEL * D_MODEL * 2);
  h16* wot  = (h16*)alloc((size_t)D_MODEL * D_MODEL * 2);
  h16* qh   = (h16*)alloc((size_t)BS * D_MODEL * 2);
  h16* khb  = (h16*)alloc((size_t)BS * D_MODEL * 2);
  h16* vhb  = (h16*)alloc((size_t)BS * D_MODEL * 2);
  h16* vtb  = (h16*)alloc((size_t)BB * N_HEADS * D_K * SS * 2);
  h16* ohb  = (h16*)alloc((size_t)BS * D_MODEL * 2);
  float* maskb = (float*)alloc((size_t)BB * N_SPECIAL * SS * 4);

  const int T = 256;
  int nx = BS * D_MODEL;
  int nw = D_MODEL * D_MODEL;

  k_cvt_f32_to_f16<<<(nx + T - 1) / T, T, 0, stream>>>(x, xh, nx);
  k_cvt_w_transpose<<<(nw + T - 1) / T, T, 0, stream>>>(wq, wqt);
  k_cvt_w_transpose<<<(nw + T - 1) / T, T, 0, stream>>>(wk, wkt);
  k_cvt_w_transpose<<<(nw + T - 1) / T, T, 0, stream>>>(wv, wvt);
  k_cvt_w_transpose<<<(nw + T - 1) / T, T, 0, stream>>>(wo, wot);

  // QKV projections: (M/16)*(N/64) = 256*16 = 4096 waves -> 1024 blocks of 128
  k_gemm_wmma<<<1024, 128, 0, stream>>>(xh, wqt, bq, qh,  nullptr, BS);
  k_gemm_wmma<<<1024, 128, 0, stream>>>(xh, wkt, bk, khb, nullptr, BS);
  k_gemm_wmma<<<1024, 128, 0, stream>>>(xh, wvt, bv, vhb, nullptr, BS);

  int nvt = BB * N_HEADS * D_K * SS;
  k_build_vt<<<(nvt + T - 1) / T, T, 0, stream>>>(vhb, vtb);
  int nm = BB * N_SPECIAL * SS;
  k_build_mask<<<(nm + T - 1) / T, T, 0, stream>>>(ids, vm, maskb);

  // attention: B * H * (S/16) = 4096 blocks
  k_attention<<<BB * N_HEADS * (SS / 16), 128, 0, stream>>>(
      qh, khb, vtb, maskb, attn_spec_out, ohb);

  // final projection, f32 out
  k_gemm_wmma<<<1024, 128, 0, stream>>>(ohb, wot, bo, nullptr, final_out, BS);
}
